// SelfAttentiveRNN_19172734009383
// MI455X (gfx1250) — matrix operand order repaired and checked by reading, not compiled, not run on hardware
//
#include <hip/hip_runtime.h>
#include <hip/hip_bf16.h>
#include <cstdint>
#include <cstddef>

// ---------------- problem constants ----------------
#define TT    512      // seq len
#define BB    32       // batch
#define EE    300      // embed
#define EP    320      // embed padded to mult of 32
#define HH    1024     // hidden
#define GG    4096     // 4*H gates
#define KC    1344     // EP + HH  (fused K for LSTM GEMM)
#define H2    2048     // 2*H
#define AD    350      // attention dim
#define ADP   384      // padded to mult of 128 (WG N-tile of blocked GEMM)
#define RR    30       // aspects
#define RP    32       // padded
#define MLPH  512
#define KMLP  61440    // RR*H2
#define CC    2
#define HTILE 32       // hidden units per LSTM workgroup

// ---------------- WMMA types ----------------
typedef __attribute__((ext_vector_type(16))) __bf16 v16bf;
typedef __attribute__((ext_vector_type(8)))  float  v8f;

union Frag16 { v16bf v; uint4 q[2]; };

__device__ __forceinline__ v8f wmma_bf16(const Frag16& a, const Frag16& b, v8f c) {
  return __builtin_amdgcn_wmma_f32_16x16x32_bf16(false, a.v, false, b.v,
                                                 (short)0, c, false, false);
}

__device__ __forceinline__ __bf16 f2bf(float f) {
  unsigned u = __builtin_bit_cast(unsigned, f);
  unsigned r = (u + 0x7FFFu + ((u >> 16) & 1u)) >> 16;
  return __builtin_bit_cast(__bf16, (unsigned short)r);
}
__device__ __forceinline__ float sigm(float x) { return 1.f / (1.f + __expf(-x)); }

// ---------------- packing kernels ----------------
__global__ void pack_embed(const int* __restrict__ inp, const float* __restrict__ Wemb,
                           __bf16* __restrict__ Xbf) {
  int idx = blockIdx.x;                 // idx = t*BB + b
  int t = idx >> 5, b = idx & 31;
  int tok = inp[b * TT + t];
  const float* er = Wemb + (long long)tok * EE;
  __bf16* dst = Xbf + (long long)idx * EP;
  for (int k = threadIdx.x; k < EP; k += blockDim.x)
    dst[k] = f2bf(k < EE ? er[k] : 0.f);
}

__global__ void pack_wcomb(const float* __restrict__ Wih, const float* __restrict__ Whh,
                           __bf16* __restrict__ Wc) {
  long long n = (long long)2 * GG * KC;
  for (long long i = (long long)blockIdx.x * blockDim.x + threadIdx.x; i < n;
       i += (long long)gridDim.x * blockDim.x) {
    int col = (int)(i % KC);
    long long rowg = i / KC;            // dir*GG + row
    int row = (int)(rowg % GG);
    int dir = (int)(rowg / GG);
    float v;
    if (col < EE)      v = Wih[((long long)dir * GG + row) * EE + col];
    else if (col < EP) v = 0.f;
    else               v = Whh[((long long)dir * GG + row) * HH + (col - EP)];
    Wc[i] = f2bf(v);
  }
}

__global__ void pack_bias(const float* bih, const float* bhh, float* bc) {
  int i = blockIdx.x * blockDim.x + threadIdx.x;
  if (i < 2 * GG) bc[i] = bih[i] + bhh[i];
}

__global__ void pack_w1(const float* __restrict__ W1, __bf16* __restrict__ W1p) {
  long long n = (long long)ADP * H2;
  for (long long i = (long long)blockIdx.x * blockDim.x + threadIdx.x; i < n;
       i += (long long)gridDim.x * blockDim.x) {
    int row = (int)(i / H2), col = (int)(i % H2);
    W1p[i] = f2bf(row < AD ? W1[(long long)row * H2 + col] : 0.f);
  }
}

__global__ void pack_w2(const float* __restrict__ W2, __bf16* __restrict__ W2p) {
  int i = blockIdx.x * blockDim.x + threadIdx.x;
  if (i < RP * ADP) {
    int row = i / ADP, col = i % ADP;
    W2p[i] = f2bf((row < RR && col < AD) ? W2[row * AD + col] : 0.f);
  }
}

__global__ void conv_bf(const float* __restrict__ src, __bf16* __restrict__ dst, long long n) {
  for (long long i = (long long)blockIdx.x * blockDim.x + threadIdx.x; i < n;
       i += (long long)gridDim.x * blockDim.x)
    dst[i] = f2bf(src[i]);
}

// ---------------- fused LSTM step (GEMM + cell update) ----------------
// grid = (HH/HTILE unit-tiles, 2 dirs), block = 256 (8 wave32).
// One workgroup owns HTILE hidden units = 4*HTILE gate columns (the same units
// of i,f,g,o), so the nonlinear cell update fuses locally through LDS.
// Weights (22 MB bf16 both dirs) stay resident in the 192 MB L2 across steps.
__global__ __launch_bounds__(256)
void lstm_step(const __bf16* __restrict__ Xbf, const __bf16* __restrict__ Wc,
               const float* __restrict__ biasc,
               const __bf16* __restrict__ h_prev, __bf16* __restrict__ h_next,
               float* __restrict__ c,
               __bf16* __restrict__ out_bf, int s) {
  __shared__ float glds[32][4 * HTILE];
  int dir = blockIdx.y;
  int t = dir ? (TT - 1 - s) : s;
  int j0 = blockIdx.x * HTILE;
  int wave = threadIdx.x >> 5, lane = threadIdx.x & 31;
  int l15 = lane & 15;
  int ako = (lane < 16) ? 0 : 8;
  int bko = (lane < 16) ? 0 : 8;

  // this wave's single N-tile: local gate col -> global Wcomb row (gate*HH + unit)
  int lc = wave * 16 + l15;                       // 0..127
  const __bf16* bp = Wc + ((long long)dir * GG + (lc >> 5) * HH + j0 + (lc & 31)) * KC;
  // A rows: concat(x_t[row][0:EP], h_prev[row][0:HH]) as two K phases
  const __bf16* x0 = Xbf + ((long long)t * BB + l15) * EP;
  const __bf16* x1 = x0 + 16 * EP;
  const __bf16* h0 = h_prev + ((long long)dir * BB + l15) * HH;
  const __bf16* h1 = h0 + 16 * HH;

  v8f acc[2] = {};
  // phase 1: K over padded embedding (EP = 320)
  for (int k0 = 0; k0 < EP; k0 += 32) {
    Frag16 a0, a1, b0;
    a0.q[0] = *(const uint4*)(x0 + k0 + ako);   a0.q[1] = *(const uint4*)(x0 + k0 + 16 + ako);
    a1.q[0] = *(const uint4*)(x1 + k0 + ako);   a1.q[1] = *(const uint4*)(x1 + k0 + 16 + ako);
    b0.q[0] = *(const uint4*)(bp + k0 + bko);   b0.q[1] = *(const uint4*)(bp + k0 + bko + 16);
    acc[0] = wmma_bf16(a0, b0, acc[0]);
    acc[1] = wmma_bf16(a1, b0, acc[1]);
  }
  // phase 2: K over recurrent h (HH = 1024), weights offset by EP
  for (int k0 = 0; k0 < HH; k0 += 32) {
    if (k0 + 512 < HH)
      __builtin_prefetch((const void*)(bp + EP + k0 + 512), 0, 1);
    Frag16 a0, a1, b0;
    a0.q[0] = *(const uint4*)(h0 + k0 + ako);        a0.q[1] = *(const uint4*)(h0 + k0 + 16 + ako);
    a1.q[0] = *(const uint4*)(h1 + k0 + ako);        a1.q[1] = *(const uint4*)(h1 + k0 + 16 + ako);
    b0.q[0] = *(const uint4*)(bp + EP + k0 + bko);   b0.q[1] = *(const uint4*)(bp + EP + k0 + bko + 16);
    acc[0] = wmma_bf16(a0, b0, acc[0]);
    acc[1] = wmma_bf16(a1, b0, acc[1]);
  }

  // spill accumulators to LDS gate buffer [32 rows][4*HTILE gate cols]
  int mb = (lane < 16) ? 0 : 8;
  for (int mi = 0; mi < 2; mi++) {
    v8f v = acc[mi];
    int col = wave * 16 + l15;
    for (int r = 0; r < 8; r++) glds[mi * 16 + mb + r][col] = v[r];
  }
  __syncthreads();

  // fused LSTM cell update (PyTorch gate order i,f,g,o)
  const float* bias = biasc + dir * GG;
  for (int e = threadIdx.x; e < 32 * HTILE; e += 256) {
    int row = e >> 5, u = e & (HTILE - 1);
    float vi = glds[row][u]              + bias[j0 + u];
    float vf = glds[row][HTILE + u]      + bias[HH + j0 + u];
    float vg = glds[row][2 * HTILE + u]  + bias[2 * HH + j0 + u];
    float vo = glds[row][3 * HTILE + u]  + bias[3 * HH + j0 + u];
    long long ci = ((long long)dir * BB + row) * HH + j0 + u;
    float cp = c[ci];
    float cn = sigm(vf) * cp + sigm(vi) * tanhf(vg);
    float hv = sigm(vo) * tanhf(cn);
    c[ci] = cn;
    __bf16 hb = f2bf(hv);
    h_next[ci] = hb;
    out_bf[((long long)row * TT + t) * H2 + dir * HH + j0 + u] = hb;  // [B][T][2H]
  }
}

// ---------------- LDS-tiled transpose: [B][T][H2] -> [B][H2][T] ----------------
// Moved off the sequential LSTM critical path; coalesced 8B loads/stores.
__global__ __launch_bounds__(256)
void transpose_out(const __bf16* __restrict__ in, __bf16* __restrict__ outT) {
  __shared__ __bf16 tile[64][72];
  int b = blockIdx.z;
  int h0 = blockIdx.x * 64, t0 = blockIdx.y * 64;
  const __bf16* src = in + (long long)b * TT * H2;
  for (int i = threadIdx.x; i < 64 * 16; i += 256) {      // 64 t-rows x 16 chunks of 4
    int r = i >> 4, cc = i & 15;
    *(uint2*)&tile[r][cc * 4] =
        *(const uint2*)(src + (long long)(t0 + r) * H2 + h0 + cc * 4);
  }
  __syncthreads();
  __bf16* dst = outT + (long long)b * H2 * TT;
  for (int i = threadIdx.x; i < 64 * 16; i += 256) {      // 64 h-rows x 16 chunks of 4
    int r = i >> 4, cc = i & 15;
    __bf16 tmp[4] = { tile[cc * 4 + 0][r], tile[cc * 4 + 1][r],
                      tile[cc * 4 + 2][r], tile[cc * 4 + 3][r] };
    *(uint2*)(dst + (long long)(h0 + r) * TT + t0 + cc * 4) = *(uint2*)tmp;
  }
}

// ---------------- blocked batched WMMA GEMM: D = epilogue(A @ B^T) --------------
// 2x2 16x16 tiles per wave (4 WMMA per 4 fragment loads -> 2x operand reuse).
// WG tile = 64(M) x 128(N); N of every call is a multiple of 128.
// flags: 1 = +bias[n], 2 = relu, 4 = tanh. Stores to Df (f32) and/or Dbf (bf16).
__global__ __launch_bounds__(256)
void gemm_bf16_wmma2x2(const __bf16* __restrict__ A, const __bf16* __restrict__ B,
                       float* __restrict__ Df, __bf16* __restrict__ Dbf,
                       const float* __restrict__ bias,
                       int M, int N, int K, int lda, int ldb, int ldd,
                       long long sA, long long sB, long long sD, int flags) {
  int z = blockIdx.z;
  A += (long long)z * sA;
  B += (long long)z * sB;
  int wave = threadIdx.x >> 5, lane = threadIdx.x & 31;
  int mg = wave & 1, ng = wave >> 1;
  int m0 = blockIdx.x * 64 + mg * 32;
  int n0 = blockIdx.y * 128 + ng * 32;
  int l15 = lane & 15;
  int rA0 = m0 + l15;       if (rA0 > M - 1) rA0 = M - 1;   // clamp for M < tile
  int rA1 = m0 + 16 + l15;  if (rA1 > M - 1) rA1 = M - 1;
  const __bf16* arow0 = A + (long long)rA0 * lda;
  const __bf16* arow1 = A + (long long)rA1 * lda;
  const __bf16* brow0 = B + (long long)(n0 + l15) * ldb;
  const __bf16* brow1 = B + (long long)(n0 + 16 + l15) * ldb;
  int ako = (lane < 16) ? 0 : 8;
  int bko = (lane < 16) ? 0 : 8;

  v8f acc[4] = {};                       // [mi][ni]
  for (int k0 = 0; k0 < K; k0 += 32) {
    if (k0 + 2048 < K) {                 // stream-prefetch the weight operand
      __builtin_prefetch((const void*)(brow0 + k0 + 2048), 0, 1);
      __builtin_prefetch((const void*)(brow1 + k0 + 2048), 0, 1);
    }
    Frag16 a0, a1, b0, b1;
    a0.q[0] = *(const uint4*)(arow0 + k0 + ako);  a0.q[1] = *(const uint4*)(arow0 + k0 + 16 + ako);
    a1.q[0] = *(const uint4*)(arow1 + k0 + ako);  a1.q[1] = *(const uint4*)(arow1 + k0 + 16 + ako);
    b0.q[0] = *(const uint4*)(brow0 + k0 + bko);  b0.q[1] = *(const uint4*)(brow0 + k0 + bko + 16);
    b1.q[0] = *(const uint4*)(brow1 + k0 + bko);  b1.q[1] = *(const uint4*)(brow1 + k0 + bko + 16);
    acc[0] = wmma_bf16(a0, b0, acc[0]);  acc[1] = wmma_bf16(a0, b1, acc[1]);
    acc[2] = wmma_bf16(a1, b0, acc[2]);  acc[3] = wmma_bf16(a1, b1, acc[3]);
  }

  for (int mi = 0; mi < 2; mi++)
    for (int ni = 0; ni < 2; ni++) {
      v8f v = acc[mi * 2 + ni];
      int mbase = m0 + mi * 16 + ((lane < 16) ? 0 : 8);
      int n = n0 + ni * 16 + l15;
      if (n < N) {
        float bv = (flags & 1) ? bias[n] : 0.f;
        for (int r = 0; r < 8; r++) {
          int m = mbase + r;
          if (m < M) {
            float x = v[r] + bv;
            if (flags & 2) x = fmaxf(x, 0.f);
            if (flags & 4) x = tanhf(x);
            long long o = (long long)z * sD + (long long)m * ldd + n;
            if (Df)  Df[o]  = x;
            if (Dbf) Dbf[o] = f2bf(x);
          }
        }
      }
    }
}

// ---------------- narrow-N WMMA GEMM (N = 32) for the aspect scores -------------
__global__ __launch_bounds__(256)
void gemm_bf16_wmma_n32(const __bf16* __restrict__ A, const __bf16* __restrict__ B,
                        float* __restrict__ Df,
                        int M, int N, int K, int lda, int ldb, int ldd) {
  int wave = threadIdx.x >> 5, lane = threadIdx.x & 31;
  int mt = wave & 3, nt = wave >> 2;
  int m0 = blockIdx.x * 64 + mt * 16;
  int n0 = nt * 16;
  int l15 = lane & 15;
  int rA = m0 + l15; if (rA > M - 1) rA = M - 1;
  const __bf16* arow = A + (long long)rA * lda;
  const __bf16* brow = B + (long long)(n0 + l15) * ldb;
  int ako = (lane < 16) ? 0 : 8;
  int bko = (lane < 16) ? 0 : 8;

  v8f acc = {};
  for (int k0 = 0; k0 < K; k0 += 32) {
    Frag16 fa, fb;
    fa.q[0] = *(const uint4*)(arow + k0 + ako);
    fa.q[1] = *(const uint4*)(arow + k0 + 16 + ako);
    fb.q[0] = *(const uint4*)(brow + k0 + bko);
    fb.q[1] = *(const uint4*)(brow + k0 + bko + 16);
    acc = wmma_bf16(fa, fb, acc);
  }
  int mbase = m0 + ((lane < 16) ? 0 : 8);
  int n = n0 + l15;
  if (n < N) {
    for (int r = 0; r < 8; r++) {
      int m = mbase + r;
      if (m < M) Df[(long long)m * ldd + n] = acc[r];
    }
  }
}

// ---------------- masked softmax over T, producing A [B][RP][T] ----------------
__global__ __launch_bounds__(256)
void softmax_attn(const float* __restrict__ s, const int* __restrict__ lens,
                  float* __restrict__ Af, __bf16* __restrict__ Abf) {
  int gw = blockIdx.x * 8 + (threadIdx.x >> 5);     // 1024 waves total
  int lane = threadIdx.x & 31;
  int b = gw >> 5, r = gw & 31;
  float*  arow = Af  + ((long long)b * RP + r) * TT;
  __bf16* brow = Abf + ((long long)b * RP + r) * TT;
  if (r >= RR) {                                     // zero padded aspect rows
    for (int t = lane; t < TT; t += 32) { arow[t] = 0.f; brow[t] = f2bf(0.f); }
    return;
  }
  int len = lens[b];
  const float* srow = s + (long long)b * TT * RP + r;   // stride RP over t
  float mx = -1e30f;
  for (int t = lane; t < TT; t += 32)
    if (t < len) mx = fmaxf(mx, srow[(long long)t * RP]);
  for (int o = 16; o; o >>= 1) mx = fmaxf(mx, __shfl_xor(mx, o, 32));
  float sum = 0.f;
  for (int t = lane; t < TT; t += 32) {
    float e = (t < len) ? __expf(srow[(long long)t * RP] - mx) : 0.f;
    arow[t] = e; sum += e;
  }
  for (int o = 16; o; o >>= 1) sum += __shfl_xor(sum, o, 32);
  float inv = 1.f / sum;
  for (int t = lane; t < TT; t += 32) {
    float v = arow[t] * inv;
    arow[t] = v; brow[t] = f2bf(v);
  }
}

// ---------------- Frobenius orthogonality penalty ----------------
__global__ __launch_bounds__(256)
void penalty_kernel(const float* __restrict__ Af, float* __restrict__ part) {
  int b = blockIdx.x;
  float acc = 0.f;
  for (int p = threadIdx.x; p < RR * RR; p += 256) {
    int r = p / RR, q = p % RR;
    const float* ar = Af + ((long long)b * RP + r) * TT;
    const float* aq = Af + ((long long)b * RP + q) * TT;
    float d = 0.f;
    for (int t = 0; t < TT; t++) d += ar[t] * aq[t];
    d -= (r == q) ? 1.f : 0.f;
    acc += d * d;
  }
  __shared__ float red[256];
  red[threadIdx.x] = acc;
  __syncthreads();
  for (int st = 128; st; st >>= 1) {
    if (threadIdx.x < st) red[threadIdx.x] += red[threadIdx.x + st];
    __syncthreads();
  }
  if (threadIdx.x == 0) part[b] = red[0];
}

__global__ void finalize_penalty(const float* part, float* out) {
  if (threadIdx.x == 0) {
    float sum = 0.f;
    for (int b = 0; b < BB; b++) sum += part[b];
    out[BB * CC] = sum / (float)BB;
  }
}

// ---------------- decoder (tiny) ----------------
__global__ void decode_kernel(const float* __restrict__ hid, const float* __restrict__ Wdec,
                              const float* __restrict__ bdec, float* __restrict__ out) {
  int b = threadIdx.x;
  if (b >= BB) return;
  float z0 = bdec[0], z1 = bdec[1];
  for (int k = 0; k < MLPH; k++) {
    float h = hid[b * MLPH + k];
    z0 += h * Wdec[k];
    z1 += h * Wdec[MLPH + k];
  }
  float m = fmaxf(z0, z1);
  float e0 = __expf(z0 - m), e1 = __expf(z1 - m);
  float inv = 1.f / (e0 + e1);
  out[b * CC]     = e0 * inv;
  out[b * CC + 1] = e1 * inv;
}

// ---------------- host orchestration ----------------
extern "C" void kernel_launch(void* const* d_in, const int* in_sizes, int n_in,
                              void* d_out, int out_size, void* d_ws, size_t ws_size,
                              hipStream_t stream) {
  const int*   inp  = (const int*)d_in[0];
  const int*   lens = (const int*)d_in[1];
  const float* Wemb = (const float*)d_in[2];
  const float* Wih  = (const float*)d_in[3];
  const float* Whh  = (const float*)d_in[4];
  const float* bih  = (const float*)d_in[5];
  const float* bhh  = (const float*)d_in[6];
  const float* W1   = (const float*)d_in[7];
  const float* W2   = (const float*)d_in[8];
  const float* Wmlp = (const float*)d_in[9];
  const float* bmlp = (const float*)d_in[10];
  const float* Wdec = (const float*)d_in[11];
  const float* bdec = (const float*)d_in[12];
  float* out = (float*)d_out;
  (void)in_sizes; (void)n_in; (void)out_size; (void)ws_size;

  char* p = (char*)d_ws;
  auto alloc = [&](size_t bytes) {
    char* r = p;
    p += (bytes + 255) & ~(size_t)255;
    return r;
  };
  __bf16* Xbf   = (__bf16*)alloc((size_t)TT * BB * EP * 2);        // 10.5 MB
  __bf16* Wc    = (__bf16*)alloc((size_t)2 * GG * KC * 2);         // 22 MB
  float*  bc    = (float*) alloc((size_t)2 * GG * 4);
  __bf16* hping = (__bf16*)alloc((size_t)2 * 2 * BB * HH * 2);     // ping-pong h
  float*  cbuf  = (float*) alloc((size_t)2 * BB * HH * 4);
  __bf16* outbf = (__bf16*)alloc((size_t)BB * TT * H2 * 2);        // 67 MB
  __bf16* outtr = (__bf16*)alloc((size_t)BB * H2 * TT * 2);        // 67 MB
  __bf16* W1p   = (__bf16*)alloc((size_t)ADP * H2 * 2);
  __bf16* tanhb = (__bf16*)alloc((size_t)BB * TT * ADP * 2);       // 12.6 MB
  __bf16* W2p   = (__bf16*)alloc((size_t)RP * ADP * 2);
  float*  sbuf  = (float*) alloc((size_t)BB * TT * RP * 4);
  float*  Af    = (float*) alloc((size_t)BB * RP * TT * 4);
  __bf16* Abf   = (__bf16*)alloc((size_t)BB * RP * TT * 2);
  __bf16* Mbf   = (__bf16*)alloc((size_t)BB * RP * H2 * 2);
  __bf16* Wmlpb = (__bf16*)alloc((size_t)MLPH * KMLP * 2);         // 63 MB
  float*  hid   = (float*) alloc((size_t)BB * MLPH * 4);
  float*  part  = (float*) alloc((size_t)BB * 4);

  hipMemsetAsync(hping, 0, (size_t)2 * 2 * BB * HH * 2, stream);   // h0 = 0
  hipMemsetAsync(cbuf,  0, (size_t)2 * BB * HH * 4, stream);       // c0 = 0

  // weight / activation packing (bf16, padded K/N)
  pack_embed<<<TT * BB, 128, 0, stream>>>(inp, Wemb, Xbf);
  pack_wcomb<<<2048, 256, 0, stream>>>(Wih, Whh, Wc);
  pack_bias<<<(2 * GG + 255) / 256, 256, 0, stream>>>(bih, bhh, bc);
  pack_w1<<<1024, 256, 0, stream>>>(W1, W1p);
  pack_w2<<<(RP * ADP + 255) / 256, 256, 0, stream>>>(W2, W2p);
  conv_bf<<<4096, 256, 0, stream>>>(Wmlp, Wmlpb, (long long)MLPH * KMLP);

  // sequential bidirectional LSTM: 512 fused GEMM+cell steps, both dirs per launch
  for (int s = 0; s < TT; s++) {
    const __bf16* hp = hping + (size_t)(s & 1) * 2 * BB * HH;
    __bf16*       hn = hping + (size_t)((s + 1) & 1) * 2 * BB * HH;
    lstm_step<<<dim3(HH / HTILE, 2), 256, 0, stream>>>(Xbf, Wc, bc, hp, hn, cbuf,
                                                       outbf, s);
  }

  // out^T for the context einsum, off the sequential critical path
  transpose_out<<<dim3(H2 / 64, TT / 64, BB), 256, 0, stream>>>(outbf, outtr);

  // s1 = tanh(out @ W1^T)  -> bf16 [B*T, ADP]
  gemm_bf16_wmma2x2<<<dim3((BB * TT) / 64, ADP / 128, 1), 256, 0, stream>>>(
      outbf, W1p, nullptr, tanhb, nullptr,
      BB * TT, ADP, H2, H2, H2, ADP, 0, 0, 0, /*tanh*/ 4);

  // s = s1 @ W2^T -> f32 [B*T, RP]
  gemm_bf16_wmma_n32<<<dim3((BB * TT) / 64, 1, 1), 256, 0, stream>>>(
      tanhb, W2p, sbuf, BB * TT, RP, ADP, ADP, ADP, RP);

  softmax_attn<<<128, 256, 0, stream>>>(sbuf, lens, Af, Abf);

  // M = A @ out (batched over B): [RP x T] @ [T x 2H] via out^T layout
  gemm_bf16_wmma2x2<<<dim3(1, H2 / 128, BB), 256, 0, stream>>>(
      Abf, outtr, nullptr, Mbf, nullptr,
      RP, H2, TT, TT, TT, H2,
      (long long)RP * TT, (long long)H2 * TT, (long long)RP * H2, 0);

  penalty_kernel<<<BB, 256, 0, stream>>>(Af, part);
  finalize_penalty<<<1, 32, 0, stream>>>(part, out);

  // hid = relu(feats @ Wmlp^T + b) ; feats row = first KMLP elems of Mbf row-block
  gemm_bf16_wmma2x2<<<dim3(1, MLPH / 128, 1), 256, 0, stream>>>(
      Mbf, Wmlpb, hid, nullptr, bmlp,
      BB, MLPH, KMLP, RP * H2, KMLP, MLPH, 0, 0, 0, /*bias|relu*/ 3);

  decode_kernel<<<1, 64, 0, stream>>>(hid, Wdec, bdec, out);
}
